// SchNetAvg_1829656068126
// MI455X (gfx1250) — matrix operand order repaired and checked by reading, not compiled, not run on hardware
//
#include <hip/hip_runtime.h>
#include <hip/hip_bf16.h>

typedef __attribute__((ext_vector_type(16))) _Float16 v16h;
typedef __attribute__((ext_vector_type(8)))  float    v8f;

#define BLOCK 256
#define WPB   8          // waves per block (wave32)
#define LOG2F_C 0.69314718055994531f

// ---------------------------------------------------------------- helpers

__device__ __forceinline__ float sspf(float x) {
  // softplus(x) - log(2), numerically stable
  return fmaxf(x, 0.0f) + log1pf(__expf(-fabsf(x))) - LOG2F_C;
}

__device__ __forceinline__ v8f wmma_f16(v16h a, v16h b, v8f c) {
  return __builtin_amdgcn_wmma_f32_16x16x32_f16(false, a, false, b, (short)0, c,
                                                false, false);
}

// B fragment: prepacked, contiguous 16 halves per lane
__device__ __forceinline__ v16h load_Bfrag(const _Float16* __restrict__ packed,
                                           int kc, int nt, int lane) {
  return *(const v16h*)(packed + (unsigned)(((kc << 2) + nt) * 32 + lane) * 16u);
}

// A fragment (16x32 f16): lane holds row M = lane&15; h = lane>>4
// element e -> K = chunk*32 + (e&7) + 8*h + 16*(e>>3)
__device__ __forceinline__ v16h load_Afrag_g(const float* __restrict__ rowp,
                                             int chunk, int h) {
  v16h a;
#pragma unroll
  for (int e = 0; e < 16; ++e) {
    int k = chunk * 32 + (e & 7) + 8 * h + ((e >> 3) << 4);
    a[e] = (_Float16)rowp[k];
  }
  return a;
}

__device__ __forceinline__ v16h load_Afrag_l(const _Float16* __restrict__ rowp,
                                             int chunk, int h) {
  v16h a;
#pragma unroll
  for (int e = 0; e < 16; ++e) {
    int k = chunk * 32 + (e & 7) + 8 * h + ((e >> 3) << 4);
    a[e] = rowp[k];
  }
  return a;
}

// within-wave LDS write->read ordering (cross-lane): drain DS counter
__device__ __forceinline__ void wave_lds_fence() {
  asm volatile("s_wait_dscnt 0x0" ::: "memory");
}

// ---------------------------------------------------------------- prepack
// Pack a Ktot x 64 row-major f32 weight into WMMA B fragments (f16).
// out[((kc*4+nt)*32+lane)*16+e] = W[kc*32 + (e&7) + 8*(lane>>4) + 16*(e>>3)][nt*16 + (lane&15)]
__global__ void pack_kernel(const float* __restrict__ W, _Float16* __restrict__ out,
                            int Ktot) {
  int t = blockIdx.x * blockDim.x + threadIdx.x;
  if (t >= 2 * 4 * 32 * 16) return;
  int e    = t & 15;
  int lane = (t >> 4) & 31;
  int f    = t >> 9;           // 0..7 : f = kc*4 + nt
  int nt   = f & 3;
  int kc   = f >> 2;
  int k = kc * 32 + (e & 7) + 8 * (lane >> 4) + ((e >> 3) << 4);
  int n = nt * 16 + (lane & 15);
  out[t] = (k < Ktot) ? (_Float16)W[k * 64 + n] : (_Float16)0.0f;
}

// ---------------------------------------------------------------- small kernels

__global__ void embed_kernel(const int* __restrict__ z, const float* __restrict__ emb,
                             float* __restrict__ X, int N) {
  int t = blockIdx.x * blockDim.x + threadIdx.x;
  if (t < N * 64) {
    int n = t >> 6, f = t & 63;
    X[t] = emb[(unsigned)z[n] * 64u + (unsigned)f];
  }
}

__global__ void solvent_kernel(const int* __restrict__ solvent,
                               const float* __restrict__ emb_s,
                               const float* __restrict__ w1, const float* __restrict__ b1,
                               const float* __restrict__ w2, const float* __restrict__ b2,
                               float* __restrict__ S) {
  __shared__ float h1[64];
  int g = blockIdx.x, j = threadIdx.x;   // 64 threads
  const float* e = emb_s + (unsigned)solvent[g] * 64u;
  float acc = b1[j];
  for (int k = 0; k < 64; ++k) acc = fmaf(e[k], w1[k * 64 + j], acc);
  h1[j] = sspf(acc);
  __syncthreads();
  if (j < 32) {
    float a2 = b2[j];
    for (int k = 0; k < 64; ++k) a2 = fmaf(h1[k], w2[k * 32 + j], a2);
    S[g * 32 + j] = a2;
  }
}

__global__ void count_kernel(const int* __restrict__ batch, float* __restrict__ cnt,
                             int N) {
  int t = blockIdx.x * blockDim.x + threadIdx.x;
  if (t < N) atomicAdd(&cnt[batch[t]], 1.0f);
}

__global__ void head_kernel(const float* __restrict__ pooled, const float* __restrict__ cnt,
                            const float* __restrict__ S,
                            const float* __restrict__ w1, const float* __restrict__ b1,
                            const float* __restrict__ w2, const float* __restrict__ b2,
                            const float* __restrict__ w3, const float* __restrict__ b3,
                            float* __restrict__ out) {
  __shared__ float in96[96];
  __shared__ float h1[128];
  __shared__ float h2[32];
  int g = blockIdx.x, j = threadIdx.x;   // 128 threads
  if (j < 64)       in96[j] = pooled[g * 64 + j] / fmaxf(cnt[g], 1.0f);
  else if (j < 96)  in96[j] = S[g * 32 + (j - 64)];
  __syncthreads();
  float a = b1[j];
  for (int k = 0; k < 96; ++k) a = fmaf(in96[k], w1[k * 128 + j], a);
  h1[j] = sspf(a);
  __syncthreads();
  if (j < 32) {
    float a2 = b2[j];
    for (int k = 0; k < 128; ++k) a2 = fmaf(h1[k], w2[k * 32 + j], a2);
    h2[j] = sspf(a2);
  }
  __syncthreads();
  if (j == 0) {
    float a3 = b3[0];
    for (int k = 0; k < 32; ++k) a3 = fmaf(h2[k], w3[k], a3);
    out[g] = a3;
  }
}

// ---------------------------------------------------------------- WMMA node GEMMs

// OUT[N,64] = IN[N,64] @ Wp + b    (one 16-row tile per wave)
__global__ void gemm1_kernel(const float* __restrict__ IN,
                             const _Float16* __restrict__ Wp, const float* __restrict__ b,
                             float* __restrict__ OUT, int Nn) {
  int lane = threadIdx.x & 31, wv = threadIdx.x >> 5;
  int tile = blockIdx.x * WPB + wv;
  int row0 = tile << 4;
  if (row0 >= Nn) return;
  int h = lane >> 4, r15 = lane & 15;
  const bool full = (row0 + 16 <= Nn);
  const float* rowp = IN + (unsigned)min(row0 + r15, Nn - 1) * 64u;
  v16h a0 = load_Afrag_g(rowp, 0, h);
  v16h a1 = load_Afrag_g(rowp, 1, h);
  unsigned base = (unsigned)(row0 + 8 * h) * 64u;
#pragma unroll
  for (int nt = 0; nt < 4; ++nt) {
    float bb = b[nt * 16 + r15];
    v8f c = {bb, bb, bb, bb, bb, bb, bb, bb};
    c = wmma_f16(a0, load_Bfrag(Wp, 0, nt, lane), c);
    c = wmma_f16(a1, load_Bfrag(Wp, 1, nt, lane), c);
    float* op = OUT + base + (unsigned)(nt * 16 + r15);
    if (full) {
#pragma unroll
      for (int r = 0; r < 8; ++r) op[r * 64] = c[r];
    } else {
#pragma unroll
      for (int r = 0; r < 8; ++r)
        if (row0 + r + 8 * h < Nn) op[r * 64] = c[r];
    }
  }
}

// 2-layer MLP: t = ssp(IN@W1+b1); o = t@W2+b2
// mode 0: XOUT += o (residual update)    mode 1: atomicAdd POOL[batch[row]] += o
__global__ void node_mlp_kernel(const float* __restrict__ IN,
                                const _Float16* __restrict__ W1p, const float* __restrict__ b1,
                                const _Float16* __restrict__ W2p, const float* __restrict__ b2,
                                float* __restrict__ XOUT, const int* __restrict__ batch,
                                float* __restrict__ POOL, int mode, int Nn) {
  __shared__ _Float16 lds[WPB][16 * 64];
  int lane = threadIdx.x & 31, wv = threadIdx.x >> 5;
  int tile = blockIdx.x * WPB + wv;
  int row0 = tile << 4;
  if (row0 >= Nn) return;
  int h = lane >> 4, r15 = lane & 15;
  const bool full = (row0 + 16 <= Nn);
  const float* rowp = IN + (unsigned)min(row0 + r15, Nn - 1) * 64u;
  v16h a0 = load_Afrag_g(rowp, 0, h);
  v16h a1 = load_Afrag_g(rowp, 1, h);
  _Float16* tl = lds[wv];
#pragma unroll
  for (int nt = 0; nt < 4; ++nt) {
    float bb = b1[nt * 16 + r15];
    v8f c = {bb, bb, bb, bb, bb, bb, bb, bb};
    c = wmma_f16(a0, load_Bfrag(W1p, 0, nt, lane), c);
    c = wmma_f16(a1, load_Bfrag(W1p, 1, nt, lane), c);
#pragma unroll
    for (int r = 0; r < 8; ++r)
      tl[(r + 8 * h) * 64 + nt * 16 + r15] = (_Float16)sspf(c[r]);
  }
  wave_lds_fence();
  const _Float16* rowl = tl + r15 * 64;
  v16h A0 = load_Afrag_l(rowl, 0, h);
  v16h A1 = load_Afrag_l(rowl, 1, h);

  unsigned base = (unsigned)(row0 + 8 * h) * 64u;
  unsigned poff[8];
  if (mode != 0) {
#pragma unroll
    for (int r = 0; r < 8; ++r)
      poff[r] = (unsigned)batch[min(row0 + r + 8 * h, Nn - 1)] * 64u;
  }
#pragma unroll
  for (int nt = 0; nt < 4; ++nt) {
    int col = nt * 16 + r15;
    float bb = b2[col];
    v8f c = {bb, bb, bb, bb, bb, bb, bb, bb};
    c = wmma_f16(A0, load_Bfrag(W2p, 0, nt, lane), c);
    c = wmma_f16(A1, load_Bfrag(W2p, 1, nt, lane), c);
    if (mode == 0) {
      float* op = XOUT + base + (unsigned)col;
      if (full) {
#pragma unroll
        for (int r = 0; r < 8; ++r) op[r * 64] += c[r];
      } else {
#pragma unroll
        for (int r = 0; r < 8; ++r)
          if (row0 + r + 8 * h < Nn) op[r * 64] += c[r];
      }
    } else {
      if (full) {
#pragma unroll
        for (int r = 0; r < 8; ++r)
          atomicAdd(&POOL[poff[r] + (unsigned)col], c[r]);
      } else {
#pragma unroll
        for (int r = 0; r < 8; ++r)
          if (row0 + r + 8 * h < Nn)
            atomicAdd(&POOL[poff[r] + (unsigned)col], c[r]);
      }
    }
  }
}

// ---------------------------------------------------------------- edge kernel
// Per wave: 16 edges. rbf -> WMMA filter MLP -> W;  msg = m[src]*W; agg[tgt] += msg
__global__ void edge_kernel(const int* __restrict__ src, const int* __restrict__ tgt,
                            const float* __restrict__ pos,
                            const _Float16* __restrict__ F1p, const float* __restrict__ f1b,
                            const _Float16* __restrict__ F2p, const float* __restrict__ f2b,
                            const float* __restrict__ Mfeat, float* __restrict__ AGG,
                            int E) {
  __shared__ _Float16 lds[WPB][16 * 64];
  int lane = threadIdx.x & 31, wv = threadIdx.x >> 5;
  int tile = blockIdx.x * WPB + wv;
  int e0 = tile << 4;
  if (e0 >= E) return;
  int h = lane >> 4, r15 = lane & 15;
  const bool full = (e0 + 16 <= E);

  // distance for this lane's row (edge e0 + r15); lanes 16-31 duplicate
  int eid = min(e0 + r15, E - 1);
  int sd = src[eid], td = tgt[eid];
  float dx = pos[sd * 3 + 0] - pos[td * 3 + 0];
  float dy = pos[sd * 3 + 1] - pos[td * 3 + 1];
  float dz = pos[sd * 3 + 2] - pos[td * 3 + 2];
  float d = sqrtf(dx * dx + dy * dy + dz * dz);

  // gaussian RBF A-fragments (K padded 51 -> 64)
  v16h a0, a1;
#pragma unroll
  for (int e = 0; e < 16; ++e) {
    int k = (e & 7) + 8 * h + ((e >> 3) << 4);
    float t0 = d - 0.1f * (float)k;
    a0[e] = (_Float16)__expf(-10.0f * t0 * t0);
    int k1 = k + 32;
    float t1 = d - 0.1f * (float)k1;
    a1[e] = (_Float16)((k1 <= 50) ? __expf(-10.0f * t1 * t1) : 0.0f);
  }

  _Float16* tl = lds[wv];
#pragma unroll
  for (int nt = 0; nt < 4; ++nt) {
    float bb = f1b[nt * 16 + r15];
    v8f c = {bb, bb, bb, bb, bb, bb, bb, bb};
    c = wmma_f16(a0, load_Bfrag(F1p, 0, nt, lane), c);
    c = wmma_f16(a1, load_Bfrag(F1p, 1, nt, lane), c);
#pragma unroll
    for (int r = 0; r < 8; ++r)
      tl[(r + 8 * h) * 64 + nt * 16 + r15] = (_Float16)sspf(c[r]);
  }
  wave_lds_fence();
  const _Float16* rowl = tl + r15 * 64;
  v16h A0 = load_Afrag_l(rowl, 0, h);
  v16h A1 = load_Afrag_l(rowl, 1, h);

  // src/tgt feature-row offsets for the 8 rows of this lane's C elements
  unsigned moff[8], aoff[8];
#pragma unroll
  for (int r = 0; r < 8; ++r) {
    int ee = min(e0 + r + 8 * h, E - 1);
    moff[r] = (unsigned)src[ee] * 64u;
    aoff[r] = (unsigned)tgt[ee] * 64u;
  }

#pragma unroll
  for (int nt = 0; nt < 4; ++nt) {
    unsigned col = (unsigned)(nt * 16 + r15);
    float bb = f2b[col];
    v8f c = {bb, bb, bb, bb, bb, bb, bb, bb};
    c = wmma_f16(A0, load_Bfrag(F2p, 0, nt, lane), c);
    c = wmma_f16(A1, load_Bfrag(F2p, 1, nt, lane), c);
    if (full) {
#pragma unroll
      for (int r = 0; r < 8; ++r) {
        float msg = c[r] * Mfeat[moff[r] + col];
        atomicAdd(&AGG[aoff[r] + col], msg);
      }
    } else {
#pragma unroll
      for (int r = 0; r < 8; ++r) {
        if (e0 + r + 8 * h < E) {
          float msg = c[r] * Mfeat[moff[r] + col];
          atomicAdd(&AGG[aoff[r] + col], msg);
        }
      }
    }
  }
}

// ---------------------------------------------------------------- launch

extern "C" void kernel_launch(void* const* d_in, const int* in_sizes, int n_in,
                              void* d_out, int out_size, void* d_ws, size_t ws_size,
                              hipStream_t stream) {
  (void)n_in; (void)out_size; (void)ws_size;
  const float* pos     = (const float*)d_in[0];
  const int*   eidx    = (const int*)d_in[1];
  const int*   z       = (const int*)d_in[2];
  const int*   batch   = (const int*)d_in[3];
  const int*   solvent = (const int*)d_in[4];
  const float* emb_z   = (const float*)d_in[5];
  const float* emb_s   = (const float*)d_in[6];
  const float* solv_w1 = (const float*)d_in[7];
  const float* solv_b1 = (const float*)d_in[8];
  const float* solv_w2 = (const float*)d_in[9];
  const float* solv_b2 = (const float*)d_in[10];
  const float* lin1_w  = (const float*)d_in[11];
  const float* lin1_b  = (const float*)d_in[12];
  const float* mlp_w1  = (const float*)d_in[13];
  const float* mlp_b1  = (const float*)d_in[14];
  const float* mlp_w2  = (const float*)d_in[15];
  const float* mlp_b2  = (const float*)d_in[16];
  const float* filt_w1 = (const float*)d_in[17];
  const float* filt_b1 = (const float*)d_in[18];
  const float* filt_w2 = (const float*)d_in[19];
  const float* filt_b2 = (const float*)d_in[20];
  const float* post_w1 = (const float*)d_in[21];
  const float* post_b1 = (const float*)d_in[22];
  const float* post_w2 = (const float*)d_in[23];
  const float* post_b2 = (const float*)d_in[24];
  const float* p2w1    = (const float*)d_in[25];
  const float* p2b1    = (const float*)d_in[26];
  const float* p2w2    = (const float*)d_in[27];
  const float* p2b2    = (const float*)d_in[28];
  const float* p2w3    = (const float*)d_in[29];
  const float* p2b3    = (const float*)d_in[30];

  const int N = in_sizes[2];
  const int E = in_sizes[1] / 2;
  const int G = in_sizes[4];
  const int* src = eidx;
  const int* tgt = eidx + E;

  char* w = (char*)d_ws;
  auto carve = [&](size_t bytes) -> char* {
    char* p = w;
    w += (bytes + 255) & ~(size_t)255;
    return p;
  };
  float*    X    = (float*)carve((size_t)N * 64 * 4);
  float*    M    = (float*)carve((size_t)N * 64 * 4);
  float*    AGG  = (float*)carve((size_t)N * 64 * 4);
  float*    S    = (float*)carve((size_t)G * 32 * 4);
  float*    POOL = (float*)carve((size_t)G * 64 * 4);
  float*    CNT  = (float*)carve((size_t)G * 4);
  _Float16* PK   = (_Float16*)carve((size_t)12 * 4096 * sizeof(_Float16));
  // slots: 0,1=lin1  2,3=filt_w1  4,5=filt_w2  6,7=mlp_w1  8,9=mlp_w2  10=post_w1 11=post_w2

  // ---- prepack weights into WMMA B-fragment layout (f16)
  for (int i = 0; i < 2; ++i) {
    pack_kernel<<<16, 256, 0, stream>>>(lin1_w  + (size_t)i * 64 * 64, PK + (size_t)(0 + i)  * 4096, 64);
    pack_kernel<<<16, 256, 0, stream>>>(filt_w1 + (size_t)i * 51 * 64, PK + (size_t)(2 + i)  * 4096, 51);
    pack_kernel<<<16, 256, 0, stream>>>(filt_w2 + (size_t)i * 64 * 64, PK + (size_t)(4 + i)  * 4096, 64);
    pack_kernel<<<16, 256, 0, stream>>>(mlp_w1  + (size_t)i * 64 * 64, PK + (size_t)(6 + i)  * 4096, 64);
    pack_kernel<<<16, 256, 0, stream>>>(mlp_w2  + (size_t)i * 64 * 64, PK + (size_t)(8 + i)  * 4096, 64);
  }
  pack_kernel<<<16, 256, 0, stream>>>(post_w1, PK + (size_t)10 * 4096, 64);
  pack_kernel<<<16, 256, 0, stream>>>(post_w2, PK + (size_t)11 * 4096, 64);

  // ---- embeddings
  embed_kernel<<<(N * 64 + 255) / 256, 256, 0, stream>>>(z, emb_z, X, N);
  solvent_kernel<<<G, 64, 0, stream>>>(solvent, emb_s, solv_w1, solv_b1, solv_w2, solv_b2, S);

  const int ntiles  = (N + 15) / 16;
  const int nblocks = (ntiles + WPB - 1) / WPB;
  const int etiles  = (E + 15) / 16;
  const int eblocks = (etiles + WPB - 1) / WPB;

  // ---- interaction blocks
  for (int i = 0; i < 2; ++i) {
    hipMemsetAsync(AGG, 0, (size_t)N * 64 * 4, stream);
    gemm1_kernel<<<nblocks, BLOCK, 0, stream>>>(X, PK + (size_t)(0 + i) * 4096,
                                                lin1_b + (size_t)i * 64, M, N);
    edge_kernel<<<eblocks, BLOCK, 0, stream>>>(src, tgt, pos,
                                               PK + (size_t)(2 + i) * 4096, filt_b1 + (size_t)i * 64,
                                               PK + (size_t)(4 + i) * 4096, filt_b2 + (size_t)i * 64,
                                               M, AGG, E);
    node_mlp_kernel<<<nblocks, BLOCK, 0, stream>>>(AGG,
                                                   PK + (size_t)(6 + i) * 4096, mlp_b1 + (size_t)i * 64,
                                                   PK + (size_t)(8 + i) * 4096, mlp_b2 + (size_t)i * 64,
                                                   X, nullptr, nullptr, /*mode=*/0, N);
  }

  // ---- post MLP + scatter-mean pooling
  hipMemsetAsync(POOL, 0, (size_t)G * 64 * 4, stream);
  hipMemsetAsync(CNT, 0, (size_t)G * 4, stream);
  count_kernel<<<(N + 255) / 256, 256, 0, stream>>>(batch, CNT, N);
  node_mlp_kernel<<<nblocks, BLOCK, 0, stream>>>(X,
                                                 PK + (size_t)10 * 4096, post_b1,
                                                 PK + (size_t)11 * 4096, post_b2,
                                                 nullptr, batch, POOL, /*mode=*/1, N);

  // ---- final head per graph
  head_kernel<<<G, 128, 0, stream>>>(POOL, CNT, S, p2w1, p2b1, p2w2, p2b2, p2w3, p2b3,
                                     (float*)d_out);
}
